// LSTM_57612691309129
// MI455X (gfx1250) — matrix operand order repaired and checked by reading, not compiled, not run on hardware
//
#include <hip/hip_runtime.h>
#include <hip/hip_bf16.h>

typedef __bf16 bf16_t;
typedef __attribute__((ext_vector_type(16))) __bf16 v16bf;
typedef __attribute__((ext_vector_type(8)))  float  v8f;

#define B_  64
#define S_  512
#define F_  128
#define H_  512
#define G_  2048   // 4*H

union V16U { v16bf v; uint4 q[2]; };
union V8FU { v8f  v; float f[8]; };

__device__ __forceinline__ float sigf(float x) { return 1.0f / (1.0f + __expf(-x)); }

// ---------------------------------------------------------------------------
// WMMA tile loaders per CDNA5 ISA VGPR layouts (CDNA5 ISA §7.12.2)
// ---------------------------------------------------------------------------

// A-tile (16x32 bf16), row-major source. Lanes 0-15: row M=lane, K {0..7} in
// v0-3 and {16..23} in v4-7. Lanes 16-31: row M=lane-16, K {8..15},{24..31}.
__device__ __forceinline__ v16bf load_a16x32(const bf16_t* __restrict__ tile,
                                             size_t row_stride, int lane) {
  const int r  = lane & 15;
  const int hi = lane >> 4;
  const bf16_t* p = tile + (size_t)r * row_stride + hi * 8;
  V16U u;
  u.q[0] = *(const uint4*)(p);        // K +0..7   (or +8..15 for hi half)
  u.q[1] = *(const uint4*)(p + 16);   // K +16..23 (or +24..31)
  return u.v;
}

// B-tile (32x16 bf16) from an N-major (pre-transposed) source: row n holds K
// contiguously. Lanes 0-15 (col n=lane): K {0..7},{16..23}; lanes 16-31: K {8..15},{24..31}.
__device__ __forceinline__ v16bf load_bT32x16(const bf16_t* __restrict__ tileT,
                                              size_t row_stride, int lane) {
  const int n  = lane & 15;
  const int hi = lane >> 4;
  const bf16_t* p = tileT + (size_t)n * row_stride + hi * 8;
  V16U u;
  u.q[0] = *(const uint4*)(p);
  u.q[1] = *(const uint4*)(p + 16);
  return u.v;
}

// ---------------------------------------------------------------------------
// Grid-wide barrier (persistent kernel, one per LSTM time step)
// ---------------------------------------------------------------------------
__device__ __forceinline__ void grid_barrier(unsigned* __restrict__ sync,
                                             unsigned nb, unsigned target) {
  __threadfence();          // release our h stores to agent scope
  __syncthreads();
  if (threadIdx.x == 0) {
    unsigned prev = __hip_atomic_fetch_add(&sync[0], 1u, __ATOMIC_ACQ_REL,
                                           __HIP_MEMORY_SCOPE_AGENT);
    if (prev == nb - 1u) {
      __hip_atomic_store(&sync[0], 0u, __ATOMIC_RELAXED, __HIP_MEMORY_SCOPE_AGENT);
      __hip_atomic_fetch_add(&sync[1], 1u, __ATOMIC_ACQ_REL, __HIP_MEMORY_SCOPE_AGENT);
    } else {
      while (__hip_atomic_load(&sync[1], __ATOMIC_ACQUIRE,
                               __HIP_MEMORY_SCOPE_AGENT) < target) {
        __builtin_amdgcn_s_sleep(2);
      }
    }
  }
  __syncthreads();
  __threadfence();          // acquire: invalidate stale cached h for all waves
}

// ---------------------------------------------------------------------------
// Weight prep
// ---------------------------------------------------------------------------
__global__ void transpose_f32_to_bf16(const float* __restrict__ src,
                                      bf16_t* __restrict__ dst, int R, int C) {
  int idx = blockIdx.x * blockDim.x + threadIdx.x;      // dst linear: c*R + r
  if (idx >= R * C) return;
  int r = idx % R;
  int c = idx / R;
  dst[idx] = (bf16_t)src[(size_t)r * C + c];
}

__global__ void convert_f32_to_bf16(const float* __restrict__ src,
                                    bf16_t* __restrict__ dst, int nelem) {
  int idx = blockIdx.x * blockDim.x + threadIdx.x;
  if (idx < nelem) dst[idx] = (bf16_t)src[idx];
}

// ---------------------------------------------------------------------------
// Persistent LSTM layer kernel.
// Grid: 32 blocks (one per 16 hidden units), 128 threads (4 waves = 4 M-tiles
// of the 64-row batch). Each wave keeps c[8] in registers for its 16x16 slice.
// Gates i,f,g,o for the same hidden unit land in the same lane -> lane-local
// cell update. h ping-pongs in global (double buffered -> 1 barrier/step).
// ---------------------------------------------------------------------------
template <int KX>
__global__ __launch_bounds__(128, 1)
void lstm_layer_kernel(const bf16_t* __restrict__ xin,   // (B, S, KX) bf16
                       const bf16_t* __restrict__ WxT,   // (G, KX) bf16 (N-major)
                       const bf16_t* __restrict__ WhT,   // (G, H)  bf16 (N-major)
                       const float*  __restrict__ bias,  // (G) f32
                       bf16_t* __restrict__ hseq,        // (B, S, H) or nullptr
                       bf16_t* __restrict__ hbuf,        // 2 * B * H ping-pong
                       unsigned* __restrict__ sync) {
  constexpr int KXP = KX + 8;   // padded LDS stride (16B aligned, bank-skewed)
  constexpr int WHP = H_ + 8;
  extern __shared__ bf16_t lds[];
  bf16_t* sWh = lds;                       // [64][WHP]
  bf16_t* sWx = lds + 64 * WHP;            // [64][KXP]

  const int tid  = threadIdx.x;
  const int lane = tid & 31;
  const int wv   = tid >> 5;               // M-tile (batch rows 16*wv..16*wv+15)
  const int j0   = blockIdx.x * 16;        // hidden-unit slice
  const int n    = lane & 15;
  const int hi   = lane >> 4;

  // Stage Wh^T slice: LDS row (g*16+jj) <- global row (g*H + j0 + jj)
  for (int i = tid; i < 64 * (H_ / 8); i += blockDim.x) {
    int nl = i / (H_ / 8);
    int u  = i % (H_ / 8);
    int ng = (nl >> 4) * H_ + j0 + (nl & 15);
    uint4 d = *((const uint4*)(WhT + (size_t)ng * H_) + u);
    *(uint4*)(sWh + (size_t)nl * WHP + u * 8) = d;
  }
  // Stage Wx^T slice
  for (int i = tid; i < 64 * (KX / 8); i += blockDim.x) {
    int nl = i / (KX / 8);
    int u  = i % (KX / 8);
    int ng = (nl >> 4) * H_ + j0 + (nl & 15);
    uint4 d = *((const uint4*)(WxT + (size_t)ng * KX) + u);
    *(uint4*)(sWx + (size_t)nl * KXP + u * 8) = d;
  }
  // Zero the t=0 read buffer (parity 0) for our hidden slice
  for (int i = tid; i < B_ * 16; i += blockDim.x) {
    int b = i >> 4;
    hbuf[(size_t)b * H_ + j0 + (i & 15)] = (bf16_t)0.0f;
  }

  const float bi = bias[0 * H_ + j0 + n];
  const float bfv = bias[1 * H_ + j0 + n];
  const float bg = bias[2 * H_ + j0 + n];
  const float bo = bias[3 * H_ + j0 + n];

  float c[8];
#pragma unroll
  for (int r = 0; r < 8; ++r) c[r] = 0.0f;

  unsigned it = 0;
  grid_barrier(sync, gridDim.x, ++it);     // LDS + h0 init visible everywhere

  const size_t in_bstride = (size_t)S_ * KX;
  const int j = j0 + n;
  const v8f vzero = {0, 0, 0, 0, 0, 0, 0, 0};

  for (int t = 0; t < S_; ++t) {
    const bf16_t* hprev = hbuf + (size_t)(t & 1) * (B_ * H_);
    bf16_t*       hcur  = hbuf + (size_t)((t + 1) & 1) * (B_ * H_);

    v8f acc[4] = {vzero, vzero, vzero, vzero};

    // Input contribution: A = x_t (16x32 per k-step), B = Wx^T gate tiles (LDS)
    const bf16_t* xbase = xin + (size_t)(wv * 16) * in_bstride + (size_t)t * KX;
#pragma unroll 4
    for (int k0 = 0; k0 < KX; k0 += 32) {
      v16bf a = load_a16x32(xbase + k0, in_bstride, lane);
#pragma unroll
      for (int g = 0; g < 4; ++g) {
        v16bf b = load_bT32x16(sWx + (size_t)(g * 16) * KXP + k0, KXP, lane);
        acc[g] = __builtin_amdgcn_wmma_f32_16x16x32_bf16(
            false, a, false, b, (short)0, acc[g], false, false);
      }
    }
    // Recurrent contribution: A = h_{t-1} (global, L2-resident), B = Wh^T (LDS)
    const bf16_t* hbase = hprev + (size_t)(wv * 16) * H_;
#pragma unroll 4
    for (int k0 = 0; k0 < H_; k0 += 32) {
      v16bf a = load_a16x32(hbase + k0, (size_t)H_, lane);
#pragma unroll
      for (int g = 0; g < 4; ++g) {
        v16bf b = load_bT32x16(sWh + (size_t)(g * 16) * WHP + k0, WHP, lane);
        acc[g] = __builtin_amdgcn_wmma_f32_16x16x32_bf16(
            false, a, false, b, (short)0, acc[g], false, false);
      }
    }

    // Lane-local LSTM cell update (C layout: M = hi*8 + r, N = lane&15)
    V8FU ui, uf, ug, uo;
    ui.v = acc[0]; uf.v = acc[1]; ug.v = acc[2]; uo.v = acc[3];
#pragma unroll
    for (int r = 0; r < 8; ++r) {
      float iv = sigf(ui.f[r] + bi);
      float fv = sigf(uf.f[r] + bfv);
      float gv = tanhf(ug.f[r] + bg);
      float ov = sigf(uo.f[r] + bo);
      c[r] = fv * c[r] + iv * gv;
      float hv = ov * tanhf(c[r]);
      const int brow = wv * 16 + hi * 8 + r;
      bf16_t hb = (bf16_t)hv;
      hcur[(size_t)brow * H_ + j] = hb;
      if (hseq) hseq[(size_t)brow * S_ * H_ + (size_t)t * H_ + j] = hb;
    }
    grid_barrier(sync, gridDim.x, ++it);
  }
}

// ---------------------------------------------------------------------------
// Dense head: out = act(A[64xK] @ W^T[NxK] + b). One block per 16 cols,
// 4 waves = 4 M-tiles. act: 0=none, 1=relu (bf16 out), 2=sigmoid (f32 out).
// ---------------------------------------------------------------------------
__global__ __launch_bounds__(128, 1)
void dense_head_kernel(const bf16_t* __restrict__ A, int K,
                       const bf16_t* __restrict__ WT,
                       const float*  __restrict__ bias,
                       bf16_t* __restrict__ outb, float* __restrict__ outf,
                       int N, int act) {
  const int lane = threadIdx.x & 31;
  const int wv   = threadIdx.x >> 5;
  const int n0   = blockIdx.x * 16;
  const int n    = lane & 15;
  const int hi   = lane >> 4;

  v8f acc = {0, 0, 0, 0, 0, 0, 0, 0};
  const bf16_t* abase = A  + (size_t)(wv * 16) * K;
  const bf16_t* wbase = WT + (size_t)n0 * K;
  for (int k0 = 0; k0 < K; k0 += 32) {
    v16bf a = load_a16x32(abase + k0, (size_t)K, lane);
    v16bf b = load_bT32x16(wbase + k0, (size_t)K, lane);
    acc = __builtin_amdgcn_wmma_f32_16x16x32_bf16(
        false, a, false, b, (short)0, acc, false, false);
  }
  V8FU u; u.v = acc;
  const float bb = bias[n0 + n];
#pragma unroll
  for (int r = 0; r < 8; ++r) {
    float v = u.f[r] + bb;
    if (act == 1) v = fmaxf(v, 0.0f);
    else if (act == 2) v = sigf(v);
    const int row = wv * 16 + hi * 8 + r;
    if (outb) outb[(size_t)row * N + n0 + n] = (bf16_t)v;
    if (outf) outf[(size_t)row * N + n0 + n] = v;
  }
}

// ---------------------------------------------------------------------------
extern "C" void kernel_launch(void* const* d_in, const int* in_sizes, int n_in,
                              void* d_out, int out_size, void* d_ws, size_t ws_size,
                              hipStream_t stream) {
  (void)in_sizes; (void)n_in; (void)out_size; (void)ws_size;
  const float* x    = (const float*)d_in[0];
  const float* Wx0  = (const float*)d_in[1];
  const float* Wh0  = (const float*)d_in[2];
  const float* b0   = (const float*)d_in[3];
  const float* Wx1  = (const float*)d_in[4];
  const float* Wh1  = (const float*)d_in[5];
  const float* b1   = (const float*)d_in[6];
  const float* Wd0  = (const float*)d_in[7];
  const float* bd0  = (const float*)d_in[8];
  const float* Wd1  = (const float*)d_in[9];
  const float* bd1  = (const float*)d_in[10];
  const float* Wout = (const float*)d_in[11];
  const float* bout = (const float*)d_in[12];

  char* ws = (char*)d_ws;
  size_t off = 0;
  auto carve = [&](size_t bytes) {
    char* p = ws + off;
    off = (off + bytes + 255) & ~(size_t)255;
    return p;
  };
  unsigned* sync  = (unsigned*)carve(256);
  bf16_t*   xb    = (bf16_t*)carve((size_t)B_ * S_ * F_ * 2);
  bf16_t*   WxT0  = (bf16_t*)carve((size_t)G_ * F_ * 2);
  bf16_t*   WhT0  = (bf16_t*)carve((size_t)G_ * H_ * 2);
  bf16_t*   WxT1  = (bf16_t*)carve((size_t)G_ * H_ * 2);
  bf16_t*   WhT1  = (bf16_t*)carve((size_t)G_ * H_ * 2);
  bf16_t*   WdT0  = (bf16_t*)carve((size_t)512 * 512 * 2);
  bf16_t*   WdT1  = (bf16_t*)carve((size_t)256 * 512 * 2);
  bf16_t*   WoutT = (bf16_t*)carve((size_t)96 * 256 * 2);
  bf16_t*   h0seq = (bf16_t*)carve((size_t)B_ * S_ * H_ * 2);
  bf16_t*   hbuf  = (bf16_t*)carve((size_t)2 * B_ * H_ * 2);
  bf16_t*   t1    = (bf16_t*)carve((size_t)B_ * 512 * 2);
  bf16_t*   t2    = (bf16_t*)carve((size_t)B_ * 256 * 2);

  const int thr = 256;
  convert_f32_to_bf16<<<(B_ * S_ * F_ + thr - 1) / thr, thr, 0, stream>>>(x, xb, B_ * S_ * F_);
  transpose_f32_to_bf16<<<(F_ * G_ + thr - 1) / thr, thr, 0, stream>>>(Wx0, WxT0, F_, G_);
  transpose_f32_to_bf16<<<(H_ * G_ + thr - 1) / thr, thr, 0, stream>>>(Wh0, WhT0, H_, G_);
  transpose_f32_to_bf16<<<(H_ * G_ + thr - 1) / thr, thr, 0, stream>>>(Wx1, WxT1, H_, G_);
  transpose_f32_to_bf16<<<(H_ * G_ + thr - 1) / thr, thr, 0, stream>>>(Wh1, WhT1, H_, G_);
  transpose_f32_to_bf16<<<(512 * 512 + thr - 1) / thr, thr, 0, stream>>>(Wd0, WdT0, 512, 512);
  transpose_f32_to_bf16<<<(512 * 256 + thr - 1) / thr, thr, 0, stream>>>(Wd1, WdT1, 512, 256);
  transpose_f32_to_bf16<<<(256 * 96 + thr - 1) / thr, thr, 0, stream>>>(Wout, WoutT, 256, 96);

  const size_t lds0 = (size_t)64 * (H_ + 8) * 2 + (size_t)64 * (F_ + 8) * 2;  //  84 KB
  const size_t lds1 = (size_t)64 * (H_ + 8) * 2 + (size_t)64 * (H_ + 8) * 2;  // 130 KB
  (void)hipFuncSetAttribute(reinterpret_cast<const void*>(&lstm_layer_kernel<F_>),
                            hipFuncAttributeMaxDynamicSharedMemorySize, (int)lds0);
  (void)hipFuncSetAttribute(reinterpret_cast<const void*>(&lstm_layer_kernel<H_>),
                            hipFuncAttributeMaxDynamicSharedMemorySize, (int)lds1);

  hipMemsetAsync(sync, 0, 256, stream);
  lstm_layer_kernel<F_><<<H_ / 16, 128, lds0, stream>>>(xb, WxT0, WhT0, b0,
                                                        h0seq, hbuf, sync);
  hipMemsetAsync(sync, 0, 256, stream);
  lstm_layer_kernel<H_><<<H_ / 16, 128, lds1, stream>>>(h0seq, WxT1, WhT1, b1,
                                                        nullptr, hbuf, sync);

  // S even -> final h of layer 1 lives in parity-0 half of hbuf.
  dense_head_kernel<<<512 / 16, 128, 0, stream>>>(hbuf, 512, WdT0, bd0, t1, nullptr, 512, 1);
  dense_head_kernel<<<256 / 16, 128, 0, stream>>>(t1,   512, WdT1, bd1, t2, nullptr, 256, 1);
  dense_head_kernel<<<96 / 16,  128, 0, stream>>>(t2,   256, WoutT, bout, nullptr,
                                                  (float*)d_out, 96, 2);
}
